// PointnetFPModuleWithImage_49469433315936
// MI455X (gfx1250) — matrix operand order repaired and checked by reading, not compiled, not run on hardware
//
#include <hip/hip_runtime.h>

typedef __attribute__((ext_vector_type(16))) _Float16 v16h;
typedef __attribute__((ext_vector_type(8)))  float    v8f;
typedef __attribute__((ext_vector_type(8)))  _Float16 h8;

// Problem constants (match reference)
constexpr int B_  = 2;
constexpr int N_  = 16384;
constexpr int M_  = 4096;
constexpr int C1_ = 64;     // unknow_feats channels
constexpr int C2_ = 256;    // known_feats channels
constexpr int C3D_ = 128;
constexpr int C2D_ = 64;
constexpr int HF_ = 96;
constexpr int WF_ = 312;
constexpr int P_  = HF_ * WF_;   // 29952, multiple of 64
constexpr int NT_ = 4;           // N-tiles per wave in the WMMA GEMM
constexpr float BN_EPS_ = 1e-5f;

// ---------------------------------------------------------------------------
// utility kernels
// ---------------------------------------------------------------------------
__global__ void k_zero_f32(float* __restrict__ p, long long n) {
  long long i = (long long)blockIdx.x * blockDim.x + threadIdx.x;
  if (i < n) p[i] = 0.0f;
}

__global__ void k_f32_to_f16(const float* __restrict__ s, _Float16* __restrict__ d, long long n) {
  long long i = (long long)blockIdx.x * blockDim.x + threadIdx.x;
  if (i < n) d[i] = (_Float16)s[i];
}

// known_feats (B,256,M) f32  ->  kft (B,M,256) f16  (row per known point)
__global__ void k_transpose_kf(const float* __restrict__ kf, _Float16* __restrict__ kft) {
  long long i = (long long)blockIdx.x * blockDim.x + threadIdx.x;
  if (i >= (long long)B_ * M_ * C2_) return;
  int c = (int)(i % C2_);
  int m = (int)((i / C2_) % M_);
  int b = (int)(i / ((long long)C2_ * M_));
  kft[i] = (_Float16)kf[((size_t)b * C2_ + c) * M_ + m];
}

// ---------------------------------------------------------------------------
// three_nn + three_interpolate + concat with unknow_feats
// writes XT1: (B, N, 320) f16 rows: [interp(256) ; unknow_feats(64)]
// known points staged in LDS as float4 (one ds_load_b128 per candidate);
// feature gather reads contiguous 512B f16 rows of kft via 16B vector loads.
// ---------------------------------------------------------------------------
__global__ void k_three_nn_interp(const float* __restrict__ unknown,      // (B,N,3)
                                  const float* __restrict__ known,        // (B,M,3)
                                  const float* __restrict__ unknow_feats, // (B,64,N)
                                  const _Float16* __restrict__ kft,       // (B,M,256)
                                  _Float16* __restrict__ xt1)             // (B,N,320)
{
  __shared__ float4 skp[M_];   // 64 KB of the 320 KB WGP LDS
  const int b   = blockIdx.y;
  const int tid = threadIdx.x;
  const int p   = blockIdx.x * 256 + tid;

  const float* kb = known + (size_t)b * M_ * 3;
  for (int i = tid; i < M_; i += 256)
    skp[i] = make_float4(kb[i * 3 + 0], kb[i * 3 + 1], kb[i * 3 + 2], 0.0f);
  __syncthreads();

  const float ux = unknown[((size_t)b * N_ + p) * 3 + 0];
  const float uy = unknown[((size_t)b * N_ + p) * 3 + 1];
  const float uz = unknown[((size_t)b * N_ + p) * 3 + 2];

  float bd0 = 3.4e38f, bd1 = 3.4e38f, bd2 = 3.4e38f;
  int   bi0 = 0, bi1 = 0, bi2 = 0;
  #pragma unroll 4
  for (int j = 0; j < M_; ++j) {
    float4 kp = skp[j];
    float dx = ux - kp.x;
    float dy = uy - kp.y;
    float dz = uz - kp.z;
    float d  = dx * dx + dy * dy + dz * dz;
    if (d < bd0)      { bd2 = bd1; bi2 = bi1; bd1 = bd0; bi1 = bi0; bd0 = d; bi0 = j; }
    else if (d < bd1) { bd2 = bd1; bi2 = bi1; bd1 = d; bi1 = j; }
    else if (d < bd2) { bd2 = d; bi2 = j; }
  }
  float w0 = 1.0f / (bd0 + 1e-8f);
  float w1 = 1.0f / (bd1 + 1e-8f);
  float w2 = 1.0f / (bd2 + 1e-8f);
  const float inv = 1.0f / (w0 + w1 + w2);
  w0 *= inv; w1 *= inv; w2 *= inv;

  const _Float16* r0 = kft + ((size_t)b * M_ + bi0) * C2_;
  const _Float16* r1 = kft + ((size_t)b * M_ + bi1) * C2_;
  const _Float16* r2 = kft + ((size_t)b * M_ + bi2) * C2_;
  _Float16* row = xt1 + ((size_t)b * N_ + p) * (C2_ + C1_);

  for (int c = 0; c < C2_; c += 8) {
    h8 a0 = *(const h8*)(r0 + c);
    h8 a1 = *(const h8*)(r1 + c);
    h8 a2 = *(const h8*)(r2 + c);
    h8 o;
    #pragma unroll
    for (int j = 0; j < 8; ++j)
      o[j] = (_Float16)(w0 * (float)a0[j] + w1 * (float)a1[j] + w2 * (float)a2[j]);
    *(h8*)(row + c) = o;
  }
  const float* uf = unknow_feats + (size_t)b * C1_ * N_;
  for (int c = 0; c < C1_; ++c)
    row[C2_ + c] = (_Float16)uf[(size_t)c * N_ + p];
}

// ---------------------------------------------------------------------------
// projection + bilinear image gather + scatter of new_features to 2D grid
// block = 128 threads, one block per (n, b)
// ---------------------------------------------------------------------------
__global__ void k_proj_gather_scatter(const float* __restrict__ unknown,   // (B,N,3)
                                      const float* __restrict__ image,     // (B,64,HF,WF)
                                      const int*   __restrict__ visArr,    // (B,N)
                                      const float* __restrict__ scaleArr,  // (B,)
                                      const float* __restrict__ rotArr,    // (B,)
                                      const int*   __restrict__ flipArr,   // (B,)
                                      const float* __restrict__ V2R,       // (B,4,4)
                                      const float* __restrict__ P2,        // (B,3,4)
                                      const int*   __restrict__ ihp,
                                      const int*   __restrict__ iwp,
                                      const float* __restrict__ NF,        // (B,128,N)
                                      _Float16* __restrict__ xt_pif,       // (B,N,64)
                                      float* __restrict__ sums,            // (B,P,128)
                                      float* __restrict__ cnt)             // (B,P)
{
  const int n = blockIdx.x;
  const int b = blockIdx.y;
  const int t = threadIdx.x;

  const float ih = (float)(*ihp);
  const float iw = (float)(*iwp);

  const float s_inv = 1.0f / scaleArr[b];
  const float a  = -rotArr[b];
  const float ca = cosf(a), sa = sinf(a);
  float x = unknown[((size_t)b * N_ + n) * 3 + 0] * s_inv;
  float y = unknown[((size_t)b * N_ + n) * 3 + 1] * s_inv;
  float z = unknown[((size_t)b * N_ + n) * 3 + 2] * s_inv;
  float xr = x * ca - y * sa;
  float yr = x * sa + y * ca;
  if (flipArr[b] != 0) yr = -yr;

  const float hom[4] = {xr, yr, z, 1.0f};
  const float* vr = V2R + (size_t)b * 16;
  float C0[4];
  #pragma unroll
  for (int e = 0; e < 4; ++e)
    C0[e] = hom[0]*vr[e*4+0] + hom[1]*vr[e*4+1] + hom[2]*vr[e*4+2] + hom[3]*vr[e*4+3];
  const float* pm = P2 + (size_t)b * 12;
  float Cc[3];
  #pragma unroll
  for (int g = 0; g < 3; ++g)
    Cc[g] = C0[0]*pm[g*4+0] + C0[1]*pm[g*4+1] + C0[2]*pm[g*4+2] + C0[3]*pm[g*4+3];

  const float u = (Cc[0] / Cc[2]) * ((float)WF_ / iw);
  const float v = (Cc[1] / Cc[2]) * ((float)HF_ / ih);
  const bool vis = visArr[(size_t)b * N_ + n] > 0;

  // bilinear gather (threads 0..63, one channel each)
  if (t < C2D_) {
    int x0 = (int)floorf(u), y0 = (int)floorf(v);
    int x0c = min(max(x0, 0), WF_ - 1), x1c = min(max(x0 + 1, 0), WF_ - 1);
    int y0c = min(max(y0, 0), HF_ - 1), y1c = min(max(y0 + 1, 0), HF_ - 1);
    float x0f = (float)x0c, x1f = (float)x1c, y0f = (float)y0c, y1f = (float)y1c;
    const float* ch = image + ((size_t)b * C2D_ + t) * (HF_ * WF_);
    float Ia = ch[y0c * WF_ + x0c];
    float Ib = ch[y1c * WF_ + x0c];
    float Ic = ch[y0c * WF_ + x1c];
    float Id = ch[y1c * WF_ + x1c];
    float wa = (x1f - u) * (y1f - v);
    float wb = (x1f - u) * (v - y0f);
    float wc = (u - x0f) * (y1f - v);
    float wd = (u - x0f) * (v - y0f);
    float val = (Ia * wa + Ib * wb + Ic * wc + Id * wd) * (vis ? 1.0f : 0.0f);
    xt_pif[((size_t)b * N_ + n) * C2D_ + t] = (_Float16)val;
  }

  // scatter to 2D grid (all 128 threads, one channel each)
  const bool valid = (u >= 0.0f) && (u < (float)WF_) && (v >= 0.0f) && (v < (float)HF_) && vis;
  if (valid) {
    int ug = (int)floorf(u), vg = (int)floorf(v);
    int ugc = min(max(ug, 0), WF_ - 1);
    int vgc = min(max(vg, 0), HF_ - 1);
    int flat = ugc * HF_ + vgc;
    float val = NF[((size_t)b * C3D_ + t) * N_ + n];
    atomicAdd(sums + ((size_t)b * P_ + flat) * C3D_ + t, val);
    if (t == 0) atomicAdd(cnt + (size_t)b * P_ + flat, 1.0f);
  }
}

// finalize scatter mean -> f16 activation (B, P, 128)
__global__ void k_finalize_scatter(const float* __restrict__ sums, const float* __restrict__ cnt,
                                   _Float16* __restrict__ xt2d) {
  long long i = (long long)blockIdx.x * blockDim.x + threadIdx.x;
  if (i >= (long long)B_ * P_ * C3D_) return;
  long long bp = i / C3D_;
  xt2d[i] = (_Float16)(sums[i] / (cnt[bp] + 1e-6f));
}

// pack image features into concat buffer rows 128..191 of (B, P, 192)
// column p = w*HF + h  (scatter grid order), image stored (c, h, w)
__global__ void k_pack_image(const float* __restrict__ image, _Float16* __restrict__ xt_cat) {
  long long i = (long long)blockIdx.x * blockDim.x + threadIdx.x;
  if (i >= (long long)B_ * C2D_ * P_) return;
  int hw = (int)(i % P_);
  int c  = (int)((i / P_) % C2D_);
  int b  = (int)(i / ((long long)P_ * C2D_));
  int h = hw / WF_, w = hw % WF_;
  int p = w * HF_ + h;
  float val = image[((size_t)b * C2D_ + c) * (HF_ * WF_) + hw];
  xt_cat[((size_t)b * P_ + p) * (C3D_ + C2D_) + C3D_ + c] = (_Float16)val;
}

// ---------------------------------------------------------------------------
// Fused WMMA GEMM:  Y(Cout,Ncols) = act(BN_or_bias( W(Cout,Cin) * X ))
// X consumed from transposed f16 activations XT: (B, Ncols, xtStride)+xtOfs
// One wave computes a 16 x (16*NT_) strip: the A (weight) fragment is loaded
// once per K-step and reused by NT_ WMMAs (4 v_wmma issue per s_wait).
// ---------------------------------------------------------------------------
template <bool USE_BN, bool OUT_F32_EN, bool OUT_XT_EN, bool REMAP_IMG>
__global__ void k_gemm_wmma_f16(const _Float16* __restrict__ W,
                                const _Float16* __restrict__ XT,
                                const float* __restrict__ bnb,   // BN (4,Cout) or bias (Cout)
                                float* __restrict__ outF32,      // (B,Cout,Ncols) [or image remap]
                                _Float16* __restrict__ outXT,    // (B,Ncols,outXtStride)+outXtOfs
                                int Cout, int Cin, int Ncols,
                                int xtStride, int xtOfs,
                                int outXtStride, int outXtOfs)
{
  const int wave = (int)(blockIdx.x * (blockDim.x >> 5) + (threadIdx.x >> 5));
  const int mtiles  = Cout >> 4;
  const int ngroups = Ncols >> 6;          // groups of NT_=4 column tiles
  const int total   = mtiles * ngroups;
  if (wave >= total) return;               // wave-uniform: EXEC stays all-ones
  const int mt = wave % mtiles;
  const int ng = wave / mtiles;
  const int b  = blockIdx.y;

  const int lane = threadIdx.x & 31;
  const int mrow = lane & 15;              // A row / output column within tile
  const int half = lane >> 4;

  const int n0 = (ng << 6) + mrow;         // first of NT_ columns handled by this lane
  const _Float16* wbase = W + (size_t)((mt << 4) + mrow) * Cin;
  const _Float16* xbase = XT + ((size_t)b * Ncols + n0) * xtStride + xtOfs;
  const size_t xtTile = (size_t)16 * xtStride;   // element stride between column tiles

  v8f acc[NT_];
  #pragma unroll
  for (int i = 0; i < NT_; ++i) acc[i] = (v8f){0.f, 0.f, 0.f, 0.f, 0.f, 0.f, 0.f, 0.f};

  for (int kb = 0; kb < Cin; kb += 32) {
    // A fragment (weights): a[j] = W[m][kb + 16*half + j], a[8+j] = W[m][kb+16*half+8+j]
    h8 w0 = *(const h8*)(wbase + kb + 16 * half);
    h8 w1 = *(const h8*)(wbase + kb + 16 * half + 8);
    v16h av;
    #pragma unroll
    for (int j = 0; j < 8; ++j) { av[j] = w0[j]; av[j + 8] = w1[j]; }
    if (kb + 32 < Cin) __builtin_prefetch(xbase + kb + 32, 0, 1);

    #pragma unroll
    for (int i = 0; i < NT_; ++i) {
      // B fragment (activations): bv[j] = X[kb + 16*half + j][n0 + i*16]
      h8 x0 = *(const h8*)(xbase + i * xtTile + kb + 16 * half);
      h8 x1 = *(const h8*)(xbase + i * xtTile + kb + 16 * half + 8);
      v16h bv;
      #pragma unroll
      for (int j = 0; j < 8; ++j) { bv[j] = x0[j]; bv[j + 8] = x1[j]; }
      acc[i] = __builtin_amdgcn_wmma_f32_16x16x32_f16(false, av, false, bv,
                                                      (short)0, acc[i], false, false);
    }
  }

  // epilogue: lane holds rows ch = mt*16 + 8*half + r (r=0..7) at columns n0 + i*16
  float sc[8], bs[8];
  #pragma unroll
  for (int r = 0; r < 8; ++r) {
    const int ch = (mt << 4) + 8 * half + r;
    if (USE_BN) {
      const float g  = bnb[ch];
      const float be = bnb[Cout + ch];
      const float mu = bnb[2 * Cout + ch];
      const float va = bnb[3 * Cout + ch];
      sc[r] = g * rsqrtf(va + BN_EPS_);
      bs[r] = be - mu * sc[r];
    } else {
      sc[r] = 1.0f;
      bs[r] = bnb[ch];
    }
  }

  #pragma unroll
  for (int i = 0; i < NT_; ++i) {
    const int n_global = n0 + (i << 4);
    float vals[8];
    #pragma unroll
    for (int r = 0; r < 8; ++r)
      vals[r] = fmaxf(acc[i][r] * sc[r] + bs[r], 0.0f);

    if (OUT_F32_EN) {
      #pragma unroll
      for (int r = 0; r < 8; ++r) {
        const int ch = (mt << 4) + 8 * half + r;
        size_t off;
        if (REMAP_IMG) {
          const int h = n_global % HF_;
          const int w = n_global / HF_;
          off = ((size_t)b * Cout + ch) * (size_t)(HF_ * WF_) + (size_t)h * WF_ + w;
        } else {
          off = ((size_t)b * Cout + ch) * (size_t)Ncols + n_global;
        }
        outF32[off] = vals[r];
      }
    }
    if (OUT_XT_EN) {
      h8 o;
      #pragma unroll
      for (int r = 0; r < 8; ++r) o[r] = (_Float16)vals[r];
      *(h8*)(outXT + ((size_t)b * Ncols + n_global) * outXtStride + outXtOfs + (mt << 4) + 8 * half) = o;
    }
  }
}

// ---------------------------------------------------------------------------
// host side
// ---------------------------------------------------------------------------
static inline size_t align256(size_t x) { return (x + 255) & ~(size_t)255; }

extern "C" void kernel_launch(void* const* d_in, const int* in_sizes, int n_in,
                              void* d_out, int out_size, void* d_ws, size_t ws_size,
                              hipStream_t stream) {
  // inputs (reference dict order)
  const float* unknown      = (const float*)d_in[0];
  const float* known        = (const float*)d_in[1];
  const float* unknow_feats = (const float*)d_in[2];
  const float* known_feats  = (const float*)d_in[3];
  const float* image        = (const float*)d_in[4];
  const int*   new_vis      = (const int*)  d_in[5];
  const float* world_scale  = (const float*)d_in[6];
  const float* world_rot    = (const float*)d_in[7];
  const int*   flip_x       = (const int*)  d_in[8];
  const float* V2R          = (const float*)d_in[9];
  const float* P2           = (const float*)d_in[10];
  const int*   image_h      = (const int*)  d_in[11];
  const int*   image_w      = (const int*)  d_in[12];
  const float* mlp_w        = (const float*)d_in[13];
  const float* mlp_bn       = (const float*)d_in[14];
  const float* f3d_pre_w    = (const float*)d_in[15];
  const float* f3d_pre_bn   = (const float*)d_in[16];
  const float* f3d_w        = (const float*)d_in[17];
  const float* f3d_bn       = (const float*)d_in[18];
  const float* f2d_mlp_w    = (const float*)d_in[19];
  const float* f2d_mlp_bn   = (const float*)d_in[20];
  const float* f2d_conv_w   = (const float*)d_in[21];
  const float* f2d_conv_b   = (const float*)d_in[22];
  (void)in_sizes; (void)n_in; (void)out_size;

  float* out = (float*)d_out;

  // ---- workspace layout ----
  char* base = (char*)d_ws;
  size_t off = 0;
  auto carve = [&](size_t bytes) { size_t o = off; off = align256(off + bytes); return o; };

  const size_t o_xt1   = carve((size_t)B_ * N_ * (C2_ + C1_) * 2);     // f16 (B,N,320)
  const size_t o_nf    = carve((size_t)B_ * C3D_ * N_ * 4);            // f32 (B,128,N)
  const size_t o_xtnf  = carve((size_t)B_ * N_ * (C3D_ + C2D_) * 2);   // f16 (B,N,192)
  const size_t o_pif   = carve((size_t)B_ * N_ * C2D_ * 2);            // f16 (B,N,64)
  const size_t o_pifa  = carve((size_t)B_ * N_ * C2D_ * 2);            // f16 (B,N,64)
  const size_t o_sums  = carve((size_t)B_ * P_ * C3D_ * 4);            // f32 (B,P,128)
  const size_t o_cnt   = carve((size_t)B_ * P_ * 4);                   // f32 (B,P)
  const size_t o_2da   = carve((size_t)B_ * P_ * C3D_ * 2);            // f16 (B,P,128)
  const size_t o_2db   = carve((size_t)B_ * P_ * C3D_ * 2);            // f16 (B,P,128)
  const size_t o_cat2d = carve((size_t)B_ * P_ * (C3D_ + C2D_) * 2);   // f16 (B,P,192)
  const size_t o_kft   = carve((size_t)B_ * M_ * C2_ * 2);             // f16 (B,M,256)
  const size_t o_w16   = carve((size_t)(C3D_*(C2_+C1_) + 3*C2D_*C2D_ + C3D_*(C3D_+C2D_)
                               + 3*C3D_*C3D_ + C2D_*(C3D_+C2D_)) * 2); // f16 weights
  if (off > ws_size) return;  // workspace too small; deterministic no-op

  _Float16* xt1    = (_Float16*)(base + o_xt1);
  float*    nf     = (float*)   (base + o_nf);
  _Float16* xtnf   = (_Float16*)(base + o_xtnf);
  _Float16* pif    = (_Float16*)(base + o_pif);
  _Float16* pifa   = (_Float16*)(base + o_pifa);
  float*    sums   = (float*)   (base + o_sums);
  float*    cnt    = (float*)   (base + o_cnt);
  _Float16* xt2da  = (_Float16*)(base + o_2da);
  _Float16* xt2db  = (_Float16*)(base + o_2db);
  _Float16* xtcat  = (_Float16*)(base + o_cat2d);
  _Float16* kft    = (_Float16*)(base + o_kft);
  _Float16* w16    = (_Float16*)(base + o_w16);

  _Float16* w_mlp  = w16;
  _Float16* w_pre  = w_mlp + (size_t)C3D_ * (C2_ + C1_);
  _Float16* w_f3d  = w_pre + (size_t)3 * C2D_ * C2D_;
  _Float16* w_2dm  = w_f3d + (size_t)C3D_ * (C3D_ + C2D_);
  _Float16* w_conv = w_2dm + (size_t)3 * C3D_ * C3D_;

  auto cvt = [&](const float* s, _Float16* d, long long n) {
    k_f32_to_f16<<<dim3((unsigned)((n + 255) / 256)), dim3(256), 0, stream>>>(s, d, n);
  };
  cvt(mlp_w,      w_mlp,  (long long)C3D_ * (C2_ + C1_));
  cvt(f3d_pre_w,  w_pre,  (long long)3 * C2D_ * C2D_);
  cvt(f3d_w,      w_f3d,  (long long)C3D_ * (C3D_ + C2D_));
  cvt(f2d_mlp_w,  w_2dm,  (long long)3 * C3D_ * C3D_);
  cvt(f2d_conv_w, w_conv, (long long)C2D_ * (C3D_ + C2D_));

  // transpose known_feats for coalesced interpolation gathers
  {
    long long n = (long long)B_ * M_ * C2_;
    k_transpose_kf<<<dim3((unsigned)((n + 255) / 256)), dim3(256), 0, stream>>>(known_feats, kft);
  }

  auto gemm_grid = [&](int Cout, int Ncols) {
    int tiles = (Cout / 16) * (Ncols / (16 * NT_));
    return dim3((unsigned)((tiles + 7) / 8), (unsigned)B_);
  };
  const dim3 gblock(256);

  // 1) three_nn + interpolate + concat -> XT1
  k_three_nn_interp<<<dim3(N_ / 256, B_), dim3(256), 0, stream>>>(
      unknown, known, unknow_feats, kft, xt1);

  // 2) mlp GEMM: (128x320) * XT1 -> NF (f32) and rows 0..127 of XTNF
  k_gemm_wmma_f16<true, true, true, false><<<gemm_grid(C3D_, N_), gblock, 0, stream>>>(
      w_mlp, xt1, mlp_bn, nf, xtnf,
      C3D_, C2_ + C1_, N_, C2_ + C1_, 0, C3D_ + C2D_, 0);

  // 3) zero scatter accumulators (must be every call: atomics accumulate)
  {
    long long ns = (long long)B_ * P_ * C3D_;
    k_zero_f32<<<dim3((unsigned)((ns + 255) / 256)), dim3(256), 0, stream>>>(sums, ns);
    long long nc = (long long)B_ * P_;
    k_zero_f32<<<dim3((unsigned)((nc + 255) / 256)), dim3(256), 0, stream>>>(cnt, nc);
  }

  // 4) projection + bilinear gather (-> pif) + scatter NF to grid (atomics)
  k_proj_gather_scatter<<<dim3(N_, B_), dim3(128), 0, stream>>>(
      unknown, image, new_vis, world_scale, world_rot, flip_x, V2R, P2,
      image_h, image_w, nf, pif, sums, cnt);

  // 5) scatter mean -> XT2D_A
  {
    long long n = (long long)B_ * P_ * C3D_;
    k_finalize_scatter<<<dim3((unsigned)((n + 255) / 256)), dim3(256), 0, stream>>>(sums, cnt, xt2da);
  }

  // 6) f2d_mlp chain: 3x (128->128) BN+ReLU, last writes rows 0..127 of XTCAT
  k_gemm_wmma_f16<true, false, true, false><<<gemm_grid(C3D_, P_), gblock, 0, stream>>>(
      w_2dm, xt2da, f2d_mlp_bn, nullptr, xt2db,
      C3D_, C3D_, P_, C3D_, 0, C3D_, 0);
  k_gemm_wmma_f16<true, false, true, false><<<gemm_grid(C3D_, P_), gblock, 0, stream>>>(
      w_2dm + (size_t)C3D_ * C3D_, xt2db, f2d_mlp_bn + 4 * C3D_, nullptr, xt2da,
      C3D_, C3D_, P_, C3D_, 0, C3D_, 0);
  k_gemm_wmma_f16<true, false, true, false><<<gemm_grid(C3D_, P_), gblock, 0, stream>>>(
      w_2dm + (size_t)2 * C3D_ * C3D_, xt2da, f2d_mlp_bn + 8 * C3D_, nullptr, xtcat,
      C3D_, C3D_, P_, C3D_, 0, C3D_ + C2D_, 0);

  // 7) pack image features into rows 128..191 of XTCAT
  {
    long long n = (long long)B_ * C2D_ * P_;
    k_pack_image<<<dim3((unsigned)((n + 255) / 256)), dim3(256), 0, stream>>>(image, xtcat);
  }

  // 8) f2d_conv GEMM: (64x192) * XTCAT, bias+ReLU -> new_img in d_out (remapped cols)
  k_gemm_wmma_f16<false, true, false, true><<<gemm_grid(C2D_, P_), gblock, 0, stream>>>(
      w_conv, xtcat, f2d_conv_b, out + (size_t)B_ * C3D_ * N_, nullptr,
      C2D_, C3D_ + C2D_, P_, C3D_ + C2D_, 0, 0, 0);

  // 9) f3d_pre chain: 3x (64->64) BN+ReLU on point image features,
  //    last writes rows 128..191 of XTNF
  k_gemm_wmma_f16<true, false, true, false><<<gemm_grid(C2D_, N_), gblock, 0, stream>>>(
      w_pre, pif, f3d_pre_bn, nullptr, pifa,
      C2D_, C2D_, N_, C2D_, 0, C2D_, 0);
  k_gemm_wmma_f16<true, false, true, false><<<gemm_grid(C2D_, N_), gblock, 0, stream>>>(
      w_pre + (size_t)C2D_ * C2D_, pifa, f3d_pre_bn + 4 * C2D_, nullptr, pif,
      C2D_, C2D_, N_, C2D_, 0, C2D_, 0);
  k_gemm_wmma_f16<true, false, true, false><<<gemm_grid(C2D_, N_), gblock, 0, stream>>>(
      w_pre + (size_t)2 * C2D_ * C2D_, pif, f3d_pre_bn + 8 * C2D_, nullptr, xtnf,
      C2D_, C2D_, N_, C2D_, 0, C3D_ + C2D_, C3D_);

  // 10) final point GEMM: (128x192) * XTNF, BN+ReLU -> new_features in d_out
  k_gemm_wmma_f16<true, true, false, false><<<gemm_grid(C3D_, N_), gblock, 0, stream>>>(
      w_f3d, xtnf, f3d_bn, out, nullptr,
      C3D_, C3D_ + C2D_, N_, C3D_ + C2D_, 0, 0, 0);
}
